// MultiHeadedAttentionRoPE_64304250355921
// MI455X (gfx1250) — compile-verified, matrix operand order checked by reading, not run on hardware
//
#include <hip/hip_runtime.h>

// ---------------------------------------------------------------------------
// MI455X (gfx1250) fused MHA+RoPE: bf16 WMMA + TDM tile staging.
// ---------------------------------------------------------------------------

typedef __attribute__((ext_vector_type(16))) __bf16 v16bf;
typedef __attribute__((ext_vector_type(8)))  __bf16 v8bf;
typedef __attribute__((ext_vector_type(8)))  float  v8f;
typedef __attribute__((ext_vector_type(4)))  float  f32x4;
typedef __attribute__((ext_vector_type(8)))  unsigned short u16x8;
typedef __attribute__((ext_vector_type(4)))  unsigned short u16x4;
typedef __attribute__((ext_vector_type(4)))  unsigned int   u32x4;
typedef __attribute__((ext_vector_type(8)))  int            i32x8;
typedef __attribute__((ext_vector_type(4)))  int            i32x4;

static constexpr int NB = 4;      // batch
static constexpr int NH = 16;     // heads
static constexpr int HD = 64;     // head dim
static constexpr int SL = 2048;   // sequence length
static constexpr int DM = 1024;   // model dim
static constexpr int MROWS = NB * SL;  // 8192

// log2(10000)/32 : RoPE inv_freq = exp2(-(dk&~1) * K)
#define ROPE_L2K 0.4152410118609203f

__device__ __forceinline__ unsigned short f2bf(float f) {
  unsigned int u = __builtin_bit_cast(unsigned int, f);
  u += 0x7fffu + ((u >> 16) & 1u);            // round-to-nearest-even
  return (unsigned short)(u >> 16);
}

__device__ __forceinline__ v16bf mkfrag(const unsigned short* p0,
                                        const unsigned short* p1) {
  v8bf lo = *(const v8bf*)p0;                 // 16B LDS/global load
  v8bf hi = *(const v8bf*)p1;                 // 16B LDS/global load
  return __builtin_shufflevector(lo, hi, 0,1,2,3,4,5,6,7,8,9,10,11,12,13,14,15);
}

__device__ __forceinline__ v8f wmma_bf16(v16bf a, v16bf b, v8f c) {
  return __builtin_amdgcn_wmma_f32_16x16x32_bf16(false, a, false, b,
                                                 (short)0, c, false, false);
}

// ---------------------------------------------------------------------------
// Tensor Data Mover: 2D bf16 tile -> LDS with +16B row padding per 128B row.
// D# built per CDNA5 ISA 8.3/8.4 (count=1, data_size=2B, type=2, pad 32dw/4dw).
// 6-arg builtin form: (g0, g1, g2, g3, g_extra, cpol).
// ---------------------------------------------------------------------------
#if __has_builtin(__builtin_amdgcn_tensor_load_to_lds)
#define USE_TDM 1
__device__ __forceinline__ unsigned lds_offset_of(const void* p) {
  return (unsigned)(unsigned long long)(const __attribute__((address_space(3))) void*)p;
}
__device__ __forceinline__ void tdm_load_2d_bf16(
    unsigned lds_off, const void* gptr,
    unsigned tensor_d0, unsigned tensor_d1,
    unsigned tile_d0, unsigned tile_d1,
    unsigned long long stride0_elems)
{
  unsigned long long ga = (unsigned long long)gptr;
  u32x4 g0;
  g0[0] = 1u;                                   // count=1, user descriptor
  g0[1] = lds_off;                              // LDS byte address
  g0[2] = (unsigned)ga;                         // global addr [31:0]
  g0[3] = (unsigned)(ga >> 32) | (2u << 30);    // global addr [56:32] | type=2
  i32x8 g1;
  unsigned w0 = (1u << 16)                      // data_size = 2 bytes
              | (1u << 20)                      // pad_enable
              | (4u << 22)                      // pad_interval: 32 DWORDs (128B)
              | (3u << 25);                     // pad_amount: 4 DWORDs (16B)
  g1[0] = (int)w0;
  g1[1] = (int)((tensor_d0 & 0xFFFFu) << 16);                       // dim0 lo
  g1[2] = (int)((tensor_d0 >> 16) | ((tensor_d1 & 0xFFFFu) << 16)); // dim0 hi|dim1 lo
  g1[3] = (int)((tensor_d1 >> 16) | (tile_d0 << 16));               // dim1 hi|tile0
  g1[4] = (int)tile_d1;                                             // tile1 (tile2=0)
  g1[5] = (int)(unsigned)(stride0_elems & 0xFFFFFFFFu);             // stride0 lo
  g1[6] = (int)(unsigned)((stride0_elems >> 32) & 0xFFFFu);         // stride0 hi
  g1[7] = 0;
  i32x4 z4 = {0, 0, 0, 0};                      // groups 2/3 unused (2D)
  i32x8 z8 = {0, 0, 0, 0, 0, 0, 0, 0};
  __builtin_amdgcn_tensor_load_to_lds(g0, g1, z4, z4, z8, 0);
}
#else
#define USE_TDM 0
#endif

// ---------------------------------------------------------------------------
// Stage 1: C = X @ W^T + bias, epilogue RoPE (+scale for Q), bf16 out.
// mode 0: Q (rope + 1/sqrt(64) scale, out [B,H,N,HD])
// mode 1: K (rope,                   out [B,H,N,HD])
// mode 2: V (plain,                  out [B,H,HD,N]  <- transposed for attn)
// ---------------------------------------------------------------------------
__global__ __launch_bounds__(256) void proj_rope_kernel(
    const float* __restrict__ X, const float* __restrict__ W,
    const float* __restrict__ bias, unsigned short* __restrict__ Out, int mode)
{
  __shared__ unsigned short As[128 * 40];   // 128 x 32 bf16, row pad +8
  __shared__ unsigned short Bs[64 * 40];    // 64  x 32 bf16, row pad +8

  const int tid  = threadIdx.x;
  const int m0   = blockIdx.x * 128;
  const int j0   = blockIdx.y * 64;
  const int wave = tid >> 5, lane = tid & 31;
  const int lr = lane & 15, lh = lane >> 4, kb = lh * 8;
  const int waveM = wave >> 1, waveN = wave & 1;

  v8f acc[2][2] = {};

  for (int kt = 0; kt < DM / 32; ++kt) {
    f32x4 aR[4];
#pragma unroll
    for (int r = 0; r < 4; ++r) {
      int row = r * 32 + (tid >> 3);
      aR[r] = *(const f32x4*)&X[(size_t)(m0 + row) * DM + kt * 32 + (tid & 7) * 4];
    }
    f32x4 bR[2];
#pragma unroll
    for (int r = 0; r < 2; ++r) {
      int row = r * 32 + (tid >> 3);
      bR[r] = *(const f32x4*)&W[(size_t)(j0 + row) * DM + kt * 32 + (tid & 7) * 4];
    }

    __syncthreads();

#pragma unroll
    for (int r = 0; r < 4; ++r) {
      int row = r * 32 + (tid >> 3);
      u16x4 c; c[0]=f2bf(aR[r][0]); c[1]=f2bf(aR[r][1]); c[2]=f2bf(aR[r][2]); c[3]=f2bf(aR[r][3]);
      *(u16x4*)&As[row * 40 + (tid & 7) * 4] = c;
    }
#pragma unroll
    for (int r = 0; r < 2; ++r) {
      int row = r * 32 + (tid >> 3);
      u16x4 c; c[0]=f2bf(bR[r][0]); c[1]=f2bf(bR[r][1]); c[2]=f2bf(bR[r][2]); c[3]=f2bf(bR[r][3]);
      *(u16x4*)&Bs[row * 40 + (tid & 7) * 4] = c;
    }

    __syncthreads();

    v16bf a[2], b[2];
#pragma unroll
    for (int mi = 0; mi < 2; ++mi) {
      const unsigned short* p = &As[(waveM * 32 + mi * 16 + lr) * 40 + kb];
      a[mi] = mkfrag(p, p + 16);
    }
#pragma unroll
    for (int ni = 0; ni < 2; ++ni) {
      const unsigned short* p = &Bs[(waveN * 32 + ni * 16 + lr) * 40 + kb];
      b[ni] = mkfrag(p, p + 16);
    }
#pragma unroll
    for (int mi = 0; mi < 2; ++mi)
#pragma unroll
      for (int ni = 0; ni < 2; ++ni)
        acc[mi][ni] = wmma_bf16(a[mi], b[ni], acc[mi][ni]);
  }

  // ---- epilogue: bias, RoPE (single v_exp for inv_freq), scale, bf16 store --
#pragma unroll
  for (int mi = 0; mi < 2; ++mi) {
#pragma unroll
    for (int ni = 0; ni < 2; ++ni) {
      int j  = j0 + waveN * 32 + ni * 16 + lr;
      int hh = j >> 6, dk = j & 63;
      float bj  = bias[j];
      float inv = 0.f;
      if (mode < 2 && dk < 32)
        inv = exp2f(-(float)(dk & ~1) * ROPE_L2K);  // THETA^(-(dk&~1)/32)
#pragma unroll
      for (int v = 0; v < 8; ++v) {
        int m  = m0 + waveM * 32 + mi * 16 + v + 8 * lh;
        int bb = m >> 11, n = m & (SL - 1);
        float x = acc[mi][ni][v] + bj;
        if (mode < 2 && dk < 32) {          // uniform over the wave
          float p = __shfl_xor(x, 1, 32);   // rotation partner (dk ^ 1)
          float ang = (float)n * inv;
          float c = __cosf(ang), s = __sinf(ang);
          x = (dk & 1) ? (x * c + p * s) : (x * c - p * s);
        }
        if (mode == 0) x *= 0.125f;         // fold 1/sqrt(HD) into Q
        size_t idx;
        if (mode == 2) idx = ((size_t)((bb * NH + hh) * HD + dk)) * SL + n;  // V^T
        else           idx = ((size_t)((bb * NH + hh) * SL + n)) * HD + dk;  // Q,K
        Out[idx] = f2bf(x);
      }
    }
  }
}

// ---------------------------------------------------------------------------
// Stage 2: flash attention per (b,h). 128 thr = 4 waves x 16 query rows.
// K/V tiles (64 keys) staged by the Tensor Data Mover into padded LDS.
// ---------------------------------------------------------------------------
__global__ __launch_bounds__(128) void attn_kernel(
    const unsigned short* __restrict__ Qb, const unsigned short* __restrict__ Kb,
    const unsigned short* __restrict__ Vb, unsigned short* __restrict__ Ctx)
{
  __shared__ unsigned short Ks[64 * 72];        // [key][dk], pad +8
  __shared__ unsigned short Vs[64 * 72];        // [dk][key], pad +8 (V^T tile)
  __shared__ unsigned short Ps[4 * 16 * 72];    // per-wave P slab [q][key]

  const int tid  = threadIdx.x;
  const int bb   = blockIdx.z, hh = blockIdx.y;
  const int q0   = blockIdx.x * 64;
  const int wave = tid >> 5, lane = tid & 31;
  const int lr = lane & 15, lh = lane >> 4, kb = lh * 8;
  const size_t head = (size_t)(bb * NH + hh);

  const unsigned short* qp = Qb + (head * SL + q0 + wave * 16 + lr) * HD;
  v16bf aq[2];
  aq[0] = mkfrag(qp + kb,      qp + kb + 16);
  aq[1] = mkfrag(qp + 32 + kb, qp + 32 + kb + 16);

  v8f m_r, l_r, o[4] = {};
#pragma unroll
  for (int i = 0; i < 8; ++i) { m_r[i] = -1e30f; l_r[i] = 0.f; }

  const unsigned short* kg = Kb + head * SL * HD;
  const unsigned short* vg = Vb + head * HD * SL;
#if USE_TDM
  const unsigned ldsK = lds_offset_of(Ks);
  const unsigned ldsV = lds_offset_of(Vs);
#else
  const int half = tid & 1;
  const int ldr  = tid >> 1;  // 0..63
#endif

  for (int kc = 0; kc < SL / 64; ++kc) {
#if USE_TDM
    __syncthreads();                       // all waves done with previous tiles
    if (wave == 0) {
      // K chunk: 64 rows x 64 dk, row-major stride HD
      tdm_load_2d_bf16(ldsK, kg + (size_t)(kc * 64) * HD, HD, SL, HD, 64, HD);
      // V^T chunk: 64 dk-rows x 64 keys, row stride SL
      tdm_load_2d_bf16(ldsV, vg + (size_t)(kc * 64),      SL, HD, 64, 64, SL);
      __builtin_amdgcn_s_wait_tensorcnt(0);
    }
    __syncthreads();                       // tiles visible to all waves
#else
    u16x8 kreg[4], vreg[4];
#pragma unroll
    for (int c = 0; c < 4; ++c) {
      kreg[c] = *(const u16x8*)&kg[(size_t)(kc * 64 + ldr) * HD + half * 32 + c * 8];
      vreg[c] = *(const u16x8*)&vg[(size_t)ldr * SL + kc * 64 + half * 32 + c * 8];
    }
    __syncthreads();
#pragma unroll
    for (int c = 0; c < 4; ++c) {
      *(u16x8*)&Ks[ldr * 72 + half * 32 + c * 8] = kreg[c];
      *(u16x8*)&Vs[ldr * 72 + half * 32 + c * 8] = vreg[c];
    }
    __syncthreads();
#endif

    // ---- S = Q K^T ----
    v8f s[4] = {};
#pragma unroll
    for (int ktl = 0; ktl < 4; ++ktl) {
      const unsigned short* kr = &Ks[(ktl * 16 + lr) * 72];
      v16bf b0 = mkfrag(kr + kb,      kr + kb + 16);
      v16bf b1 = mkfrag(kr + 32 + kb, kr + 32 + kb + 16);
      s[ktl] = wmma_bf16(aq[0], b0, s[ktl]);
      s[ktl] = wmma_bf16(aq[1], b1, s[ktl]);
    }

    // ---- online softmax ----
    v8f mx, sc;
#pragma unroll
    for (int i = 0; i < 8; ++i) {
      float v = fmaxf(fmaxf(s[0][i], s[1][i]), fmaxf(s[2][i], s[3][i]));
      v = fmaxf(v, __shfl_xor(v, 1, 32));
      v = fmaxf(v, __shfl_xor(v, 2, 32));
      v = fmaxf(v, __shfl_xor(v, 4, 32));
      v = fmaxf(v, __shfl_xor(v, 8, 32));
      mx[i] = fmaxf(v, m_r[i]);
      sc[i] = __expf(m_r[i] - mx[i]);
      m_r[i] = mx[i];
    }

    v8f su;
#pragma unroll
    for (int i = 0; i < 8; ++i) su[i] = 0.f;
#pragma unroll
    for (int ktl = 0; ktl < 4; ++ktl) {
#pragma unroll
      for (int i = 0; i < 8; ++i) {
        float p = __expf(s[ktl][i] - m_r[i]);
        su[i] += p;
        Ps[wave * 16 * 72 + (i + 8 * lh) * 72 + ktl * 16 + lr] = f2bf(p);
      }
    }
#pragma unroll
    for (int i = 0; i < 8; ++i) {
      float v = su[i];
      v += __shfl_xor(v, 1, 32);
      v += __shfl_xor(v, 2, 32);
      v += __shfl_xor(v, 4, 32);
      v += __shfl_xor(v, 8, 32);
      l_r[i] = l_r[i] * sc[i] + v;
    }
#pragma unroll
    for (int dt = 0; dt < 4; ++dt)
#pragma unroll
      for (int i = 0; i < 8; ++i) o[dt][i] *= sc[i];

    // ---- O += P V (per-wave LDS slab, in-order DS) ----
    const unsigned short* pr = &Ps[wave * 16 * 72 + lr * 72];
    v16bf ap0 = mkfrag(pr + kb,      pr + kb + 16);
    v16bf ap1 = mkfrag(pr + 32 + kb, pr + 32 + kb + 16);
#pragma unroll
    for (int dt = 0; dt < 4; ++dt) {
      const unsigned short* vr = &Vs[(dt * 16 + lr) * 72];
      v16bf bv0 = mkfrag(vr + kb,      vr + kb + 16);
      v16bf bv1 = mkfrag(vr + 32 + kb, vr + 32 + kb + 16);
      o[dt] = wmma_bf16(ap0, bv0, o[dt]);
      o[dt] = wmma_bf16(ap1, bv1, o[dt]);
    }
  }

  // ---- normalize and store context bf16 [B, N, NH*HD] ----
#pragma unroll
  for (int dt = 0; dt < 4; ++dt) {
    int dk = dt * 16 + lr;
#pragma unroll
    for (int i = 0; i < 8; ++i) {
      int n = q0 + wave * 16 + i + 8 * lh;
      float val = o[dt][i] / l_r[i];
      Ctx[((size_t)(bb * SL + n)) * DM + hh * HD + dk] = f2bf(val);
    }
  }
}

// ---------------------------------------------------------------------------
// Stage 3: out = Ctx(bf16) @ Wo^T + bo  -> fp32.
// ---------------------------------------------------------------------------
__global__ __launch_bounds__(256) void oproj_kernel(
    const unsigned short* __restrict__ Xc, const float* __restrict__ W,
    const float* __restrict__ bias, float* __restrict__ Out)
{
  __shared__ unsigned short As[128 * 40];
  __shared__ unsigned short Bs[64 * 40];

  const int tid  = threadIdx.x;
  const int m0   = blockIdx.x * 128;
  const int j0   = blockIdx.y * 64;
  const int wave = tid >> 5, lane = tid & 31;
  const int lr = lane & 15, lh = lane >> 4, kb = lh * 8;
  const int waveM = wave >> 1, waveN = wave & 1;

  v8f acc[2][2] = {};

  for (int kt = 0; kt < DM / 32; ++kt) {
    u16x8 aR[2];
#pragma unroll
    for (int r = 0; r < 2; ++r) {
      int row = r * 64 + (tid >> 2);
      aR[r] = *(const u16x8*)&Xc[(size_t)(m0 + row) * DM + kt * 32 + (tid & 3) * 8];
    }
    f32x4 bR[2];
#pragma unroll
    for (int r = 0; r < 2; ++r) {
      int row = r * 32 + (tid >> 3);
      bR[r] = *(const f32x4*)&W[(size_t)(j0 + row) * DM + kt * 32 + (tid & 7) * 4];
    }

    __syncthreads();
#pragma unroll
    for (int r = 0; r < 2; ++r) {
      int row = r * 64 + (tid >> 2);
      *(u16x8*)&As[row * 40 + (tid & 3) * 8] = aR[r];
    }
#pragma unroll
    for (int r = 0; r < 2; ++r) {
      int row = r * 32 + (tid >> 3);
      u16x4 c; c[0]=f2bf(bR[r][0]); c[1]=f2bf(bR[r][1]); c[2]=f2bf(bR[r][2]); c[3]=f2bf(bR[r][3]);
      *(u16x4*)&Bs[row * 40 + (tid & 7) * 4] = c;
    }
    __syncthreads();

    v16bf a[2], b[2];
#pragma unroll
    for (int mi = 0; mi < 2; ++mi) {
      const unsigned short* p = &As[(waveM * 32 + mi * 16 + lr) * 40 + kb];
      a[mi] = mkfrag(p, p + 16);
    }
#pragma unroll
    for (int ni = 0; ni < 2; ++ni) {
      const unsigned short* p = &Bs[(waveN * 32 + ni * 16 + lr) * 40 + kb];
      b[ni] = mkfrag(p, p + 16);
    }
#pragma unroll
    for (int mi = 0; mi < 2; ++mi)
#pragma unroll
      for (int ni = 0; ni < 2; ++ni)
        acc[mi][ni] = wmma_bf16(a[mi], b[ni], acc[mi][ni]);
  }

#pragma unroll
  for (int mi = 0; mi < 2; ++mi) {
#pragma unroll
    for (int ni = 0; ni < 2; ++ni) {
      int j = j0 + waveN * 32 + ni * 16 + lr;
      float bj = bias[j];
#pragma unroll
      for (int v = 0; v < 8; ++v) {
        int m = m0 + waveM * 32 + mi * 16 + v + 8 * lh;
        Out[(size_t)m * DM + j] = acc[mi][ni][v] + bj;
      }
    }
  }
}

// ---------------------------------------------------------------------------
extern "C" void kernel_launch(void* const* d_in, const int* in_sizes, int n_in,
                              void* d_out, int out_size, void* d_ws, size_t ws_size,
                              hipStream_t stream) {
  const float* query = (const float*)d_in[0];
  const float* key   = (const float*)d_in[1];
  const float* value = (const float*)d_in[2];
  const float* Wq = (const float*)d_in[3]; const float* bq = (const float*)d_in[4];
  const float* Wk = (const float*)d_in[5]; const float* bk = (const float*)d_in[6];
  const float* Wv = (const float*)d_in[7]; const float* bv = (const float*)d_in[8];
  const float* Wo = (const float*)d_in[9]; const float* bo = (const float*)d_in[10];
  float* out = (float*)d_out;

  unsigned short* ws = (unsigned short*)d_ws;
  const size_t HE = (size_t)NB * NH * SL * HD;   // 8M bf16 elements (16 MB)
  unsigned short* Qb = ws;
  unsigned short* Kb = ws + HE;
  unsigned short* Vb = ws + 2 * HE;
  unsigned short* Cx = ws + 3 * HE;

  dim3 gp(MROWS / 128, DM / 64);                 // 64 x 16 blocks
  proj_rope_kernel<<<gp, 256, 0, stream>>>(query, Wq, bq, Qb, 0);
  proj_rope_kernel<<<gp, 256, 0, stream>>>(key,   Wk, bk, Kb, 1);
  proj_rope_kernel<<<gp, 256, 0, stream>>>(value, Wv, bv, Vb, 2);
  attn_kernel<<<dim3(SL / 64, NH, NB), 128, 0, stream>>>(Qb, Kb, Vb, Cx);
  oproj_kernel<<<gp, 256, 0, stream>>>(Cx, Wo, bo, out);
}
